// MEPOML_GAT_83451214561529
// MI455X (gfx1250) — matrix operand order repaired and checked by reading, not compiled, not run on hardware
//
#include <hip/hip_runtime.h>
#include <hip/hip_bf16.h>
#include <stdint.h>

// ---------------------------------------------------------------------------
// Types for CDNA5 WMMA (gfx1250, wave32)
// ---------------------------------------------------------------------------
typedef __attribute__((ext_vector_type(16))) __bf16 v16bf;
typedef __attribute__((ext_vector_type(8)))  float  v8f;
typedef __attribute__((ext_vector_type(4)))  int    v4i;

typedef __attribute__((address_space(1))) v4i* gp128_t;   // global int4*
typedef __attribute__((address_space(3))) v4i* lp128_t;   // LDS int4*

union BFU { unsigned short u; __bf16 h; };

__device__ __forceinline__ unsigned short f2bf(float f) {
    unsigned u = __float_as_uint(f);
    unsigned lsb = (u >> 16) & 1u;
    u += 0x7fffu + lsb;
    return (unsigned short)(u >> 16);
}

__device__ __forceinline__ unsigned fenc(float f) {
    unsigned u = __float_as_uint(f);
    return (u & 0x80000000u) ? ~u : (u | 0x80000000u);
}
__device__ __forceinline__ float fdec(unsigned e) {
    unsigned v = (e & 0x80000000u) ? (e & 0x7fffffffu) : ~e;
    return __uint_as_float(v);
}

#define MPAD   10112      // 79 * 128, >= 10000
#define NNODES 10000
#define NHEADS 20
#define NCHAN  35
#define HIDP   704

// ---------------------------------------------------------------------------
// Async global->LDS copy support (gfx1250). Fallback: synchronous copy.
// ---------------------------------------------------------------------------
#if defined(__has_builtin)
#if __has_builtin(__builtin_amdgcn_global_load_async_to_lds_b128)
#define USE_ASYNC_LDS 1
#endif
#endif
#ifndef USE_ASYNC_LDS
#define USE_ASYNC_LDS 0
#endif

#if __has_builtin(__builtin_amdgcn_s_wait_asynccnt)
#define WAIT_ASYNC(n) __builtin_amdgcn_s_wait_asynccnt(n)
#else
#define WAIT_ASYNC(n) asm volatile("s_wait_asynccnt %0" ::"i"(n) : "memory")
#endif

__device__ __forceinline__ void cp16_async_or_sync(const unsigned short* g, unsigned short* l) {
#if USE_ASYNC_LDS
    __builtin_amdgcn_global_load_async_to_lds_b128((gp128_t)g, (lp128_t)l, 0, 0);
#else
    *(uint4*)l = *(const uint4*)g;
#endif
}

// ---------------------------------------------------------------------------
// WMMA GEMM:  C[MPAD x Npad] = A[MPAD x Kpad](bf16) * W^T, with
//   BT = weights pre-transposed, zero-padded: [NP64 x Kpad] (row n = output col)
// block tile 128(M) x 64(N), 8 waves each computing 32x32 (2x2 wmma tiles),
// LDS double-buffered via GLOBAL_LOAD_ASYNC_TO_LDS_B128 + s_wait_asynccnt.
// ---------------------------------------------------------------------------
__global__ __launch_bounds__(256)
void gemm_bf16_wmma(const unsigned short* __restrict__ A,
                    const unsigned short* __restrict__ BT,
                    const float* __restrict__ bias,
                    float* __restrict__ Cmat,
                    int Kpad, int Npad, int Nreal)
{
    __shared__ unsigned short As[2][128 * 40];   // stride 40 kills bank conflicts
    __shared__ unsigned short Bs[2][64 * 40];    // [col][k]

    const int tid  = threadIdx.x;
    const int wave = tid >> 5;
    const int lane = tid & 31;
    const int half = lane >> 4;
    const int idx  = lane & 15;
    const int wm   = wave >> 1;     // 0..3
    const int wn   = wave & 1;      // 0..1

    const long m0 = (long)blockIdx.y * 128;
    const int  n0 = blockIdx.x * 64;

    // per-thread staging chunks (all 16B-aligned, no guards: buffers are padded)
    const int arow = tid >> 1;            // 0..127
    const int aseg = (tid & 1) * 16;      // 0 or 16
    const int bcol = tid >> 2;            // 0..63
    const int bseg = (tid & 3) * 8;       // 0,8,16,24
    const unsigned short* agbase = A  + (m0 + arow) * (long)Kpad + aseg;
    const unsigned short* bgbase = BT + (long)(n0 + bcol) * Kpad + bseg;
    unsigned short* al0 = &As[0][arow * 40 + aseg];
    unsigned short* al1 = &As[1][arow * 40 + aseg];
    unsigned short* bl0 = &Bs[0][bcol * 40 + bseg];
    unsigned short* bl1 = &Bs[1][bcol * 40 + bseg];

    v8f acc[2][2] = {};

    // prologue: stage k0 = 0 into buffer 0
    cp16_async_or_sync(agbase, al0);
    cp16_async_or_sync(agbase + 8, al0 + 8);
    cp16_async_or_sync(bgbase, bl0);

    int buf = 0;
    for (int k0 = 0; k0 < Kpad; k0 += 32, buf ^= 1) {
        const bool has_next = (k0 + 32 < Kpad);
        if (has_next) {
            // issue next stage into the other buffer (3 async ops / wave)
            unsigned short* an = buf ? al0 : al1;
            unsigned short* bn = buf ? bl0 : bl1;
            cp16_async_or_sync(agbase + k0 + 32, an);
            cp16_async_or_sync(agbase + k0 + 40, an + 8);
            cp16_async_or_sync(bgbase + k0 + 32, bn);
#if USE_ASYNC_LDS
            WAIT_ASYNC(3);   // current stage complete; next stage may be in flight
#endif
        } else {
#if USE_ASYNC_LDS
            WAIT_ASYNC(0);
#endif
        }
        __syncthreads();

        const unsigned short* Ab = As[buf];
        const unsigned short* Bb = Bs[buf];

        // fragments per ISA 7.12.2 layouts
        v16bf af[2], bfr[2];
        #pragma unroll
        for (int t = 0; t < 2; ++t) {
            const int ar = wm * 32 + t * 16 + idx;
            #pragma unroll
            for (int j = 0; j < 16; ++j) {
                const int ko = ((j < 8) ? j : j + 8) + 8 * half;
                BFU pv; pv.u = Ab[ar * 40 + ko];
                af[t][j] = pv.h;
            }
            const int bc = wn * 32 + t * 16 + idx;
            #pragma unroll
            for (int j = 0; j < 16; ++j) {
                BFU pv; pv.u = Bb[bc * 40 + ((j < 8) ? j : j + 8) + 8 * half];
                bfr[t][j] = pv.h;
            }
        }

        acc[0][0] = __builtin_amdgcn_wmma_f32_16x16x32_bf16(false, af[0], false, bfr[0], (short)0, acc[0][0], false, false);
        acc[0][1] = __builtin_amdgcn_wmma_f32_16x16x32_bf16(false, af[0], false, bfr[1], (short)0, acc[0][1], false, false);
        acc[1][0] = __builtin_amdgcn_wmma_f32_16x16x32_bf16(false, af[1], false, bfr[0], (short)0, acc[1][0], false, false);
        acc[1][1] = __builtin_amdgcn_wmma_f32_16x16x32_bf16(false, af[1], false, bfr[1], (short)0, acc[1][1], false, false);

        __syncthreads();   // compute done everywhere before next stage overwrites buf^1
    }

    // epilogue: C/D layout (VGPR r: lanes0-15 -> M=r, lanes16-31 -> M=r+8)
    #pragma unroll
    for (int tm = 0; tm < 2; ++tm) {
        #pragma unroll
        for (int tn = 0; tn < 2; ++tn) {
            const int col = n0 + wn * 32 + tn * 16 + idx;
            if (col < Npad) {
                const float bv = (col < Nreal) ? bias[col] : 0.0f;
                const long rowb = m0 + wm * 32 + tm * 16 + 8 * half;
                #pragma unroll
                for (int r = 0; r < 8; ++r)
                    Cmat[(rowb + r) * (long)Npad + col] = acc[tm][tn][r] + bv;
            }
        }
    }
}

// ---------------------------------------------------------------------------
// Utility kernels
// ---------------------------------------------------------------------------
__global__ void zero_k(float* p, long n) {
    long i = blockIdx.x * (long)blockDim.x + threadIdx.x;
    if (i < n) p[i] = 0.0f;
}

// fp32 [rr x cr] -> bf16 padded [* x cp]
__global__ void cvt_pad_k(const float* __restrict__ src, unsigned short* __restrict__ dst,
                          int rr, int cr, int cp, long total) {
    long i = blockIdx.x * (long)blockDim.x + threadIdx.x;
    if (i >= total) return;
    long r = i / cp; int c = (int)(i % cp);
    float v = (r < rr && c < cr) ? src[r * (long)cr + c] : 0.0f;
    dst[i] = f2bf(v);
}

// weights fp32 [Kreal x Nreal] -> bf16 TRANSPOSED padded [NP64 x Kpad]
__global__ void cvt_wt_k(const float* __restrict__ W, unsigned short* __restrict__ WT,
                         int Kreal, int Nreal, int Kp, long total) {
    long i = blockIdx.x * (long)blockDim.x + threadIdx.x;
    if (i >= total) return;
    long n = i / Kp; int k = (int)(i % Kp);
    float v = (k < Kreal && n < Nreal) ? W[(long)k * Nreal + n] : 0.0f;
    WT[i] = f2bf(v);
}

// fp32 padded [* x Npad] -> bf16 same shape (pad cols zeroed); no BN/ReLU
__global__ void cvt_same_k(const float* __restrict__ src, unsigned short* __restrict__ dst,
                           int Npad, int Nreal, long total) {
    long i = blockIdx.x * (long)blockDim.x + threadIdx.x;
    if (i >= total) return;
    int c = (int)(i % Npad);
    float v = (c < Nreal) ? src[i] : 0.0f;
    dst[i] = f2bf(v);
}

__global__ void bias_add_k(float* __restrict__ X, const float* __restrict__ b,
                           int Npad, int Nreal, long total) {
    long i = blockIdx.x * (long)blockDim.x + threadIdx.x;
    if (i >= total) return;
    int c = (int)(i % Npad);
    if (c < Nreal) X[i] += b[c];
}

// per-column mean/var over first NNODES rows (training-mode BN, biased var)
__global__ void bn_stats_k(const float* __restrict__ X, int Npad,
                           float* __restrict__ mean, float* __restrict__ var) {
    __shared__ float s1[256], s2[256];
    const int c = blockIdx.x;
    float a = 0.0f, b = 0.0f;
    for (int r = threadIdx.x; r < NNODES; r += 256) {
        float v = X[(long)r * Npad + c];
        a += v; b += v * v;
    }
    s1[threadIdx.x] = a; s2[threadIdx.x] = b;
    __syncthreads();
    for (int o = 128; o > 0; o >>= 1) {
        if (threadIdx.x < o) { s1[threadIdx.x] += s1[threadIdx.x + o]; s2[threadIdx.x] += s2[threadIdx.x + o]; }
        __syncthreads();
    }
    if (threadIdx.x == 0) {
        float m = s1[0] * (1.0f / NNODES);
        mean[c] = m;
        var[c]  = s2[0] * (1.0f / NNODES) - m * m;
    }
}

// BN normalize + ReLU, output bf16 activation (pad cols zeroed)
__global__ void bn_norm_k(const float* __restrict__ X,
                          const float* __restrict__ mean, const float* __restrict__ var,
                          const float* __restrict__ g, const float* __restrict__ bt,
                          unsigned short* __restrict__ A, int Npad, int Nreal, long total) {
    long i = blockIdx.x * (long)blockDim.x + threadIdx.x;
    if (i >= total) return;
    int c = (int)(i % Npad);
    float v = 0.0f;
    if (c < Nreal) {
        v = (X[i] - mean[c]) * rsqrtf(var[c] + 1e-5f) * g[c] + bt[c];
        v = v > 0.0f ? v : 0.0f;
    }
    A[i] = f2bf(v);
}

// ---------------------------------------------------------------------------
// GATv2 edge-phase kernels (edge e in [0,EE): e<E -> graph edge, else self loop)
// ---------------------------------------------------------------------------
__global__ void gat_score_k(const int* __restrict__ ei, int E, int EE,
                            const float* __restrict__ XL, const float* __restrict__ XR,
                            const float* __restrict__ att,
                            float* __restrict__ score, unsigned* __restrict__ smax) {
    int t = blockIdx.x * blockDim.x + threadIdx.x;
    if (t >= EE * NHEADS) return;
    int e = t / NHEADS, h = t % NHEADS;
    int s, d;
    if (e < E) { s = ei[e]; d = ei[E + e]; } else { s = d = e - E; }
    const float* pl = XL + (long)s * HIDP + h * NCHAN;
    const float* pr = XR + (long)d * HIDP + h * NCHAN;
    const float* pa = att + h * NCHAN;
    float acc = 0.0f;
    #pragma unroll
    for (int c = 0; c < NCHAN; ++c) {
        float v = pl[c] + pr[c];
        v = v > 0.0f ? v : 0.2f * v;     // leaky_relu(0.2)
        acc += pa[c] * v;
    }
    score[t] = acc;
    atomicMax(&smax[d * NHEADS + h], fenc(acc));
}

__global__ void gat_exp_k(const int* __restrict__ ei, int E, int EE,
                          const float* __restrict__ score, const unsigned* __restrict__ smax,
                          float* __restrict__ exb, float* __restrict__ denom) {
    int t = blockIdx.x * blockDim.x + threadIdx.x;
    if (t >= EE * NHEADS) return;
    int e = t / NHEADS, h = t % NHEADS;
    int d = (e < E) ? ei[E + e] : (e - E);
    float m  = fdec(smax[d * NHEADS + h]);
    float ex = __expf(score[t] - m);
    exb[t] = ex;
    atomicAdd(&denom[d * NHEADS + h], ex);
}

__global__ void gat_agg_k(const int* __restrict__ ei, int E, int EE,
                          const float* __restrict__ XL,
                          const float* __restrict__ exb, const float* __restrict__ denom,
                          float* __restrict__ agg) {
    int t = blockIdx.x * blockDim.x + threadIdx.x;
    if (t >= EE * NHEADS) return;
    int e = t / NHEADS, h = t % NHEADS;
    int s, d;
    if (e < E) { s = ei[e]; d = ei[E + e]; } else { s = d = e - E; }
    float alpha = exb[t] / (denom[d * NHEADS + h] + 1e-16f);
    const float* pl = XL + (long)s * HIDP + h * NCHAN;
    float* po = agg + (long)d * HIDP + h * NCHAN;
    #pragma unroll
    for (int c = 0; c < NCHAN; ++c)
        atomicAdd(&po[c], pl[c] * alpha);
}

// ---------------------------------------------------------------------------
// Output head + deterministic mean subtraction
// ---------------------------------------------------------------------------
__global__ void head_k(const float* __restrict__ X, const float* __restrict__ W,
                       const float* __restrict__ b, float* __restrict__ y, int n) {
    int i = blockIdx.x * blockDim.x + threadIdx.x;
    if (i >= n) return;
    float a = b[0];
    #pragma unroll
    for (int j = 0; j < 32; ++j) a += X[(long)i * 32 + j] * W[j];
    y[i] = a;
}

__global__ void rsum_k(const float* __restrict__ y, int n, float* __restrict__ part) {
    __shared__ float s[256];
    float a = 0.0f;
    for (int i = blockIdx.x * 256 + threadIdx.x; i < n; i += gridDim.x * 256) a += y[i];
    s[threadIdx.x] = a; __syncthreads();
    for (int o = 128; o > 0; o >>= 1) {
        if (threadIdx.x < o) s[threadIdx.x] += s[threadIdx.x + o];
        __syncthreads();
    }
    if (threadIdx.x == 0) part[blockIdx.x] = s[0];
}

__global__ void fin_k(const float* __restrict__ part, int nb, float* __restrict__ mv, float invn) {
    __shared__ float s[64];
    int t = threadIdx.x;
    s[t] = (t < nb) ? part[t] : 0.0f; __syncthreads();
    for (int o = 32; o > 0; o >>= 1) {
        if (t < o) s[t] += s[t + o];
        __syncthreads();
    }
    if (t == 0) mv[0] = s[0] * invn;
}

__global__ void sub_k(const float* __restrict__ y, const float* __restrict__ mv,
                      float* __restrict__ out, int n) {
    int i = blockIdx.x * blockDim.x + threadIdx.x;
    if (i < n) out[i] = y[i] - mv[0];
}

// ---------------------------------------------------------------------------
// Host orchestration
// ---------------------------------------------------------------------------
static inline long cdivl(long a, long b) { return (a + b - 1) / b; }

extern "C" void kernel_launch(void* const* d_in, const int* in_sizes, int n_in,
                              void* d_out, int out_size, void* d_ws, size_t ws_size,
                              hipStream_t stream) {
    (void)n_in; (void)out_size; (void)ws_size;

    const int N = NNODES, IN = 226, MID1 = 463, HID = 700, MID2 = 366, OUTD = 32;
    const int P_IN = 256, P1 = 480, PH = HIDP, P2 = 384, PO = 32;
    const int LAYERS = 7;

    const float* x  = (const float*)d_in[0];
    const int*   ei = (const int*)d_in[1];
    const int E  = in_sizes[1] / 2;
    const int EE = E + N;

    // ---- carve workspace ----
    char* wp = (char*)d_ws;
    auto carve = [&](size_t bytes) -> void* {
        void* r = (void*)wp;
        wp += (bytes + 255) & ~(size_t)255;
        return r;
    };
    unsigned short* Abf  = (unsigned short*)carve((size_t)MPAD * PH * 2);
    unsigned short* Wbf  = (unsigned short*)carve((size_t)PH * PH * 2);  // fits max NP64*Kpad
    float* Cb    = (float*)carve((size_t)MPAD * PH * 4);
    float* XL    = (float*)carve((size_t)MPAD * PH * 4);
    float* XR    = (float*)carve((size_t)MPAD * PH * 4);
    float* score = (float*)carve((size_t)EE * NHEADS * 4);
    float* exb   = (float*)carve((size_t)EE * NHEADS * 4);
    unsigned* smax = (unsigned*)carve((size_t)N * NHEADS * 4);
    float* denom = (float*)carve((size_t)N * NHEADS * 4);
    float* meanb = (float*)carve((size_t)PH * 4);
    float* varb  = (float*)carve((size_t)PH * 4);
    float* ybuf  = (float*)carve((size_t)MPAD * 4);
    float* part  = (float*)carve(64 * 4);
    float* meanv = (float*)carve(4);
    float* AGG   = Cb;

    const int TPB = 256;
    auto gblk = [&](long n) { return dim3((unsigned)cdivl(n, TPB)); };

    auto run_gemm = [&](const unsigned short* Ain, const float* W, const float* b,
                        float* Cout, int Kreal, int Nreal, int Kp, int Np) {
        const int NP64 = (int)cdivl(Np, 64) * 64;
        const long tot = (long)NP64 * Kp;
        cvt_wt_k<<<gblk(tot), TPB, 0, stream>>>(W, Wbf, Kreal, Nreal, Kp, tot);
        dim3 grid((unsigned)(NP64 / 64), MPAD / 128);
        gemm_bf16_wmma<<<grid, 256, 0, stream>>>(Ain, Wbf, b, Cout, Kp, Np, Nreal);
    };
    auto run_bn = [&](const float* X, const float* g, const float* bt, int Nreal, int Np) {
        bn_stats_k<<<Nreal, 256, 0, stream>>>(X, Np, meanb, varb);
        bn_norm_k<<<gblk((long)MPAD * Np), TPB, 0, stream>>>(X, meanb, varb, g, bt, Abf, Np, Nreal, (long)MPAD * Np);
    };

    // param leaf indices (setup_inputs insertion order, depth-first)
    const int GAT0 = 10;
    const int OUT0 = GAT0 + 6 * 8 + 6;   // 64

    // ---- input MLP ----
    cvt_pad_k<<<gblk((long)MPAD * P_IN), TPB, 0, stream>>>(x, Abf, N, IN, P_IN, (long)MPAD * P_IN);
    run_gemm(Abf, (const float*)d_in[2], (const float*)d_in[3], Cb, IN, MID1, P_IN, P1);
    run_bn(Cb, (const float*)d_in[4], (const float*)d_in[5], MID1, P1);
    run_gemm(Abf, (const float*)d_in[6], (const float*)d_in[7], Cb, MID1, HID, P1, PH);
    run_bn(Cb, (const float*)d_in[8], (const float*)d_in[9], HID, PH);

    // ---- 7 GATv2 layers ----
    for (int l = 0; l < LAYERS; ++l) {
        const int base = GAT0 + 8 * l;
        const float* Wl   = (const float*)d_in[base + 0];
        const float* bl   = (const float*)d_in[base + 1];
        const float* Wr   = (const float*)d_in[base + 2];
        const float* br   = (const float*)d_in[base + 3];
        const float* att  = (const float*)d_in[base + 4];
        const float* bias = (const float*)d_in[base + 5];

        run_gemm(Abf, Wl, bl, XL, HID, HID, PH, PH);
        run_gemm(Abf, Wr, br, XR, HID, HID, PH, PH);

        zero_k<<<gblk((long)N * NHEADS), TPB, 0, stream>>>((float*)smax, (long)N * NHEADS);
        zero_k<<<gblk((long)N * NHEADS), TPB, 0, stream>>>(denom, (long)N * NHEADS);
        zero_k<<<gblk((long)MPAD * PH), TPB, 0, stream>>>(AGG, (long)MPAD * PH);

        const long T = (long)EE * NHEADS;
        gat_score_k<<<gblk(T), TPB, 0, stream>>>(ei, E, EE, XL, XR, att, score, smax);
        gat_exp_k  <<<gblk(T), TPB, 0, stream>>>(ei, E, EE, score, smax, exb, denom);
        gat_agg_k  <<<gblk(T), TPB, 0, stream>>>(ei, E, EE, XL, exb, denom, AGG);

        bias_add_k<<<gblk((long)MPAD * PH), TPB, 0, stream>>>(AGG, bias, PH, HID, (long)MPAD * PH);

        if (l < LAYERS - 1) {
            run_bn(AGG, (const float*)d_in[base + 6], (const float*)d_in[base + 7], HID, PH);
        } else {
            cvt_same_k<<<gblk((long)MPAD * PH), TPB, 0, stream>>>(AGG, Abf, PH, HID, (long)MPAD * PH);
        }
    }

    // ---- output MLP ----
    run_gemm(Abf, (const float*)d_in[OUT0 + 0], (const float*)d_in[OUT0 + 1], Cb, HID, MID2, PH, P2);
    run_bn(Cb, (const float*)d_in[OUT0 + 2], (const float*)d_in[OUT0 + 3], MID2, P2);
    run_gemm(Abf, (const float*)d_in[OUT0 + 4], (const float*)d_in[OUT0 + 5], Cb, MID2, OUTD, P2, PO);

    // ---- final linear + mean subtraction ----
    head_k<<<gblk(N), TPB, 0, stream>>>(Cb, (const float*)d_in[OUT0 + 6], (const float*)d_in[OUT0 + 7], ybuf, N);
    rsum_k<<<64, 256, 0, stream>>>(ybuf, N, part);
    fin_k<<<1, 64, 0, stream>>>(part, 64, meanv, 1.0f / N);
    sub_k<<<gblk(N), TPB, 0, stream>>>(ybuf, meanv, (float*)d_out, N);
}
